// SFM_68719476736264
// MI455X (gfx1250) — matrix operand order repaired and checked by reading, not compiled, not run on hardware
//
#include <hip/hip_runtime.h>

typedef __attribute__((ext_vector_type(16))) _Float16 v16h;
typedef __attribute__((ext_vector_type(8)))  float    v8f;

#define DT   0.2f
#define EPSC 1e-8f

// ---- raw-ISA helpers (avoid IEEE fixup/canonicalize expansions) ----
__device__ __forceinline__ float sqrt_raw(float x) { return __builtin_amdgcn_sqrtf(x); }
__device__ __forceinline__ float rcp_raw(float x)  { return __builtin_amdgcn_rcpf(x); }
__device__ __forceinline__ float max_raw(float a, float b) {
    float r;
    asm("v_max_num_f32 %0, %1, %2" : "=v"(r) : "v"(a), "v"(b));
    return r;
}
// pack two f32 -> 2xf16 and relu them in exactly two VALU ops
__device__ __forceinline__ unsigned relu_cvt_pk(float a, float b) {
    unsigned p, r;
    asm("v_cvt_pk_f16_f32 %0, %1, %2" : "=v"(p) : "v"(a), "v"(b));
    asm("v_pk_max_num_f16 %0, %1, 0"  : "=v"(r) : "v"(p));
    return r;
}

union HPack { unsigned u[8]; v16h v; };

struct MlpW {
    v16h a0;   // hidden chunk 0: row m -> {-wi[m], -bi[m]} at K=0,1
    v16h a1;   // hidden chunk 1: row m -> {-wi[m+16], -bi[m+16]}
    v16h a2;   // output layer: wo broadcast rows; lanes<16: wo[0:8]+wo[16:24], lanes>=16: wo[8:16]+wo[24:32]
    float bo;
};

__device__ __forceinline__ MlpW load_mlp(const float* __restrict__ wi,
                                         const float* __restrict__ bi,
                                         const float* __restrict__ wo,
                                         const float* __restrict__ bo,
                                         int lane) {
    MlpW W;
    float wn  = -wi[lane];
    float bn  = -bi[lane];
    float wnh = __shfl_xor(wn, 16, 32);
    float bnh = __shfl_xor(bn, 16, 32);
    // Lanes >=16 of A map to K rows 8-15 / 24-31; B lanes >=16 hold x and 1 at
    // K=8,9, so those A entries also enter the product; no lane masking here.
    v16h a0 = {}; v16h a1 = {};
    a0[0] = (_Float16)wn;  a0[1] = (_Float16)bn;
    a1[0] = (_Float16)wnh; a1[1] = (_Float16)bnh;
    const float* wb = wo + 8 * (lane >> 4);
    v16h a2;
#pragma unroll
    for (int i = 0; i < 16; ++i) a2[i] = (_Float16)wb[(i & 7) + 16 * (i >> 3)];
    W.a0 = a0; W.a1 = a1; W.a2 = a2; W.bo = bo[0];
    return W;
}

// Wave-cooperative MLP: out = wo . relu(-(x*wi + bi)) + bo for all 32 per-lane
// inputs x; each lane gets its own result. 6 WMMAs per call.
__device__ __forceinline__ float mlp_eval(const MlpW& W, float x, int lane,
                                          v16h& bA, v16h& bB) {
    bool lo = lane < 16;
    float xs = __shfl_xor(x, 16, 32);
    bA[0] = (_Float16)x;    // K=0 row for agents 0-15 (lanes 16-31: K=8 row for agents 16-31)
    bB[0] = (_Float16)xs;   // K=0 row for agents 16-31
    v8f z = {};
    v8f d0A = __builtin_amdgcn_wmma_f32_16x16x32_f16(false, W.a0, false, bA, (short)0, z, false, false);
    v8f d1A = __builtin_amdgcn_wmma_f32_16x16x32_f16(false, W.a1, false, bA, (short)0, z, false, false);
    v8f d0B = __builtin_amdgcn_wmma_f32_16x16x32_f16(false, W.a0, false, bB, (short)0, z, false, false);
    v8f d1B = __builtin_amdgcn_wmma_f32_16x16x32_f16(false, W.a1, false, bB, (short)0, z, false, false);
    HPack hA, hB;
#pragma unroll
    for (int i = 0; i < 4; ++i) {
        hA.u[i]     = relu_cvt_pk(d0A[2 * i], d0A[2 * i + 1]);
        hA.u[i + 4] = relu_cvt_pk(d1A[2 * i], d1A[2 * i + 1]);
        hB.u[i]     = relu_cvt_pk(d0B[2 * i], d0B[2 * i + 1]);
        hB.u[i + 4] = relu_cvt_pk(d1B[2 * i], d1B[2 * i + 1]);
    }
    v8f oA = __builtin_amdgcn_wmma_f32_16x16x32_f16(false, W.a2, false, hA.v, (short)0, z, false, false);
    v8f oB = __builtin_amdgcn_wmma_f32_16x16x32_f16(false, W.a2, false, hB.v, (short)0, z, false, false);
    return (lo ? oA[0] : oB[0]) + W.bo;
}

__device__ __forceinline__ void clamp_acc(float fx, float fy, float ex, float ey,
                                          float na, float ang, float& ax, float& ay) {
    float nb = max_raw(sqrt_raw(fx * fx + fy * fy), EPSC);
    float c  = (ex * fx + ey * fy) * rcp_raw(na * nb);
    bool keep = fabsf(c) > ang;
    ax += keep ? fx : 0.f;
    ay += keep ? fy : 0.f;
}

__global__ __launch_bounds__(256) void sfm_kernel(
    const float* __restrict__ ego, const float* __restrict__ nei,
    const float* __restrict__ border, const float* __restrict__ adp,
    const float* __restrict__ effang,
    const float* __restrict__ an_wi, const float* __restrict__ an_bi,
    const float* __restrict__ an_wo, const float* __restrict__ an_bo,
    const float* __restrict__ rn_wi, const float* __restrict__ rn_bi,
    const float* __restrict__ rn_wo, const float* __restrict__ rn_bo,
    const float* __restrict__ rb_wi, const float* __restrict__ rb_bi,
    const float* __restrict__ rb_wo, const float* __restrict__ rb_bo,
    const float* __restrict__ dl_wi, const float* __restrict__ dl_bi,
    const float* __restrict__ dl_wo, const float* __restrict__ dl_bo,
    float* __restrict__ out) {

    int g    = blockIdx.x * 256 + threadIdx.x;   // one agent per lane
    int lane = threadIdx.x & 31;

    // Persistent WMMA B-operand buffers: elem1 = 1 (bias row of the rank-2
    // hidden product); everything else stays zero for the whole kernel.
    v16h bA = {}; bA[1] = (_Float16)1.f;
    v16h bB = bA;

    MlpW AN = load_mlp(an_wi, an_bi, an_wo, an_bo, lane);
    MlpW RN = load_mlp(rn_wi, rn_bi, rn_wo, rn_bo, lane);
    MlpW RB = load_mlp(rb_wi, rb_bi, rb_wo, rb_bo, lane);

    // dur + dl MLP: batch-invariant (depends only on ego[1]); lane holds dl(dur[lane&7])
    float dl_vals;
    {
        MlpW DL = load_mlp(dl_wi, dl_bi, dl_wo, dl_bo, lane);
        int k = lane & 7;
        float idv = ego[93 + k];                 // ego[1, 2, 8+k]
        float dur = 0.f;
#pragma unroll
        for (int t = 0; t < 3; ++t) {
            bool f = false;
#pragma unroll
            for (int j = 0; j < 8; ++j) f = f || (idv == ego[59 + 17 * t + j]);
            dur += f ? 1.f : 0.f;
        }
        dl_vals = mlp_eval(DL, dur, lane, bA, bB);
    }

    // ego_last fields 0..5
    const float* eg = ego + (size_t)g * 51 + 34;
    float f0 = eg[0], f1 = eg[1], f2 = eg[2], f3 = eg[3], f4 = eg[4], f5 = eg[5];
    float p0 = adp[0], p1 = adp[1], ang = effang[0];
    float rcp_p0 = rcp_raw(p0);

    // e = normalize(ego_last[:,4:6]) (no eps, as in reference)
    float rcp_en = rcp_raw(sqrt_raw(f4 * f4 + f5 * f5));
    float ex = f4 * rcp_en, ey = f5 * rcp_en;
    float na = max_raw(sqrt_raw(ex * ex + ey * ey), EPSC);

    float ax = 0.f, ay = 0.f;

    // f_dest: dv = [|v34|, 0], v34 = fields 3:5
    float dvn = sqrt_raw(f3 * f3 + f4 * f4);
    clamp_acc((p1 * dvn - f3) * rcp_p0, (-f4) * rcp_p0, ex, ey, na, ang, ax, ay);

    // neighbors
    const float* nb0 = nei + (size_t)g * 136;
#pragma unroll
    for (int k = 0; k < 8; ++k) {
        const float* nb = nb0 + k * 17;
        float nx = nb[2], ny = nb[3], vx = nb[4], vy = nb[5];
        float rx = nx - f2, ry = ny - f3;
        float rnm = sqrt_raw(rx * rx + ry * ry);
        float rcp_rnm = rcp_raw(rnm);

        float m_an = mlp_eval(AN, rnm, lane, bA, bB);
        float dlk  = __shfl(dl_vals, k, 32);
        float sA   = m_an * dlk * rcp_rnm;
        clamp_acc((nx == 0.f) ? 0.f : rx * sA,
                  (ny == 0.f) ? 0.f : ry * sA, ex, ey, na, ang, ax, ay);

        float tx = rx + vx * DT, ty = ry + vy * DT;
        float bb = rnm + (tx * tx + ty * ty) - (vx * vx + vy * vy) * (DT * DT);
        float bx = sqrt_raw(bb) * 0.5f;

        float m_rn = mlp_eval(RN, bx, lane, bA, bB);
        float sR   = m_rn * rcp_rnm;
        clamp_acc((nx == 0.f) ? 0.f : rx * sR,
                  (ny == 0.f) ? 0.f : ry * sR, ex, ey, na, ang, ax, ay);
    }

    // borders: indices 0 and 3; force vector is (0, mono(|rbv|)*sign(rbv))
    {
        float rbv0 = f3 - border[0];
        float rbv1 = f3 - border[3];
        float rbn0 = fabsf(rbv0);
        float rbn1 = fabsf(rbv1);
        float m0 = mlp_eval(RB, rbn0, lane, bA, bB);
        float m1 = mlp_eval(RB, rbn1, lane, bA, bB);
        clamp_acc(0.f, m0 * copysignf(1.f, rbv0), ex, ey, na, ang, ax, ay);
        clamp_acc(0.f, m1 * copysignf(1.f, rbv1), ex, ey, na, ang, ax, ay);
    }

    float vxo = f2 + ax * DT, vyo = f3 + ay * DT;
    float sx = f0 + vxo * DT, sy = f1 + vyo * DT;

    float2* o = (float2*)(out + (size_t)g * 6);
    o[0] = make_float2(sx, sy);
    o[1] = make_float2(vxo, vyo);
    o[2] = make_float2(ax, ay);
}

extern "C" void kernel_launch(void* const* d_in, const int* in_sizes, int n_in,
                              void* d_out, int out_size, void* d_ws, size_t ws_size,
                              hipStream_t stream) {
    const float* ego    = (const float*)d_in[0];
    const float* nei    = (const float*)d_in[1];
    const float* border = (const float*)d_in[2];
    const float* adp    = (const float*)d_in[3];
    const float* effang = (const float*)d_in[4];
    const float* an_wi = (const float*)d_in[5],  *an_bi = (const float*)d_in[6];
    const float* an_wo = (const float*)d_in[7],  *an_bo = (const float*)d_in[8];
    const float* rn_wi = (const float*)d_in[9],  *rn_bi = (const float*)d_in[10];
    const float* rn_wo = (const float*)d_in[11], *rn_bo = (const float*)d_in[12];
    const float* rb_wi = (const float*)d_in[13], *rb_bi = (const float*)d_in[14];
    const float* rb_wo = (const float*)d_in[15], *rb_bo = (const float*)d_in[16];
    const float* dl_wi = (const float*)d_in[17], *dl_bi = (const float*)d_in[18];
    const float* dl_wo = (const float*)d_in[19], *dl_bo = (const float*)d_in[20];
    float* out = (float*)d_out;

    int B = in_sizes[0] / 51;                 // ego is (B, 3, 17)
    int grid = B / 256;                        // B = 262144 -> 1024 full blocks
    sfm_kernel<<<grid, 256, 0, stream>>>(ego, nei, border, adp, effang,
                                         an_wi, an_bi, an_wo, an_bo,
                                         rn_wi, rn_bi, rn_wo, rn_bo,
                                         rb_wi, rb_bi, rb_wo, rb_bo,
                                         dl_wi, dl_bi, dl_wo, dl_bo, out);
}
